// RecurrentRGCN_73624329388179
// MI455X (gfx1250) — compile-verified, matrix operand order checked by reading, not run on hardware
//
#include <hip/hip_runtime.h>
#include <hip/hip_bf16.h>

#define NENT 100000
#define DIM  128
#define NREL 500
#define RPAD 512
#define TSTEPS 3
#define NEDGE 250000
#define NSUB  500000
#define RRELU_SLOPE 0.2291666666666667f

typedef __attribute__((ext_vector_type(16))) __bf16 v16bf;
typedef __attribute__((ext_vector_type(8)))  float  v8f;

union BF16Frag { v16bf v; unsigned short h[16]; uint4 q[2]; };

__device__ __forceinline__ unsigned short f2bf_bits(float x) {
  unsigned u = __float_as_uint(x);
  unsigned r = u + 0x7FFFu + ((u >> 16) & 1u);   // round-to-nearest-even
  return (unsigned short)(r >> 16);
}

__device__ __forceinline__ float wave_sum(float v) {
#pragma unroll
  for (int off = 16; off > 0; off >>= 1) v += __shfl_xor(v, off, 32);
  return v;
}
__device__ __forceinline__ float wave_max(float v) {
#pragma unroll
  for (int off = 16; off > 0; off >>= 1) v = fmaxf(v, __shfl_xor(v, off, 32));
  return v;
}

// ---------------- WMMA matmul: C[M,ncols] = Abf[M,128] @ Bt^T ----------------
// A: bf16 row-major (pre-converted), K=128.
// Bt: bf16 [n][k] layout (row n = output column n, 128 contiguous k's).
// One wave per 16(M) x 64(N) slab: 4 accumulators share each A fragment.
// Fragments are double-buffered so K-step (s+1) loads overlap K-step (s) WMMAs.
__global__ void mm_bf16_wmma(const unsigned short* __restrict__ A,
                             const unsigned short* __restrict__ Bt,
                             float* __restrict__ C, int mtiles, int ldc) {
  const int lane  = threadIdx.x & 31;
  const int wid   = threadIdx.x >> 5;
  const int mtile = blockIdx.x * 8 + wid;
  if (mtile >= mtiles) return;                  // wave-uniform branch, EXEC stays full
  const int mbase = mtile * 16;
  const int nbase = blockIdx.y * 64;
  const int koff  = (lane >> 4) * 8;            // lanes 0-15: K 0..7/16..23; 16-31: 8..15/24..31
  const unsigned short* Arow = A + (size_t)(mbase + (lane & 15)) * DIM;
  const unsigned short* Bp   = Bt + (size_t)(nbase + (lane & 15)) * DIM;

  BF16Frag a[2], b[2][4];
  v8f acc[4];
#pragma unroll
  for (int j = 0; j < 4; ++j) acc[j] = (v8f){0.f, 0.f, 0.f, 0.f, 0.f, 0.f, 0.f, 0.f};

  // prologue: load K-step 0 into buffer 0
  a[0].q[0] = *(const uint4*)(Arow + koff);
  a[0].q[1] = *(const uint4*)(Arow + 16 + koff);
#pragma unroll
  for (int j = 0; j < 4; ++j) {
    b[0][j].q[0] = *(const uint4*)(Bp + j * 16 * DIM + koff);
    b[0][j].q[1] = *(const uint4*)(Bp + j * 16 * DIM + 16 + koff);
  }
#pragma unroll
  for (int s = 0; s < 4; ++s) {
    const int cur = s & 1, nxt = cur ^ 1;
    if (s < 3) {                                // prefetch next K-step into other buffer
      const int kb = (s + 1) * 32;
      a[nxt].q[0] = *(const uint4*)(Arow + kb + koff);
      a[nxt].q[1] = *(const uint4*)(Arow + kb + 16 + koff);
#pragma unroll
      for (int j = 0; j < 4; ++j) {
        b[nxt][j].q[0] = *(const uint4*)(Bp + j * 16 * DIM + kb + koff);
        b[nxt][j].q[1] = *(const uint4*)(Bp + j * 16 * DIM + kb + 16 + koff);
      }
    }
#pragma unroll
    for (int j = 0; j < 4; ++j)
      acc[j] = __builtin_amdgcn_wmma_f32_16x16x32_bf16(false, a[cur].v, false, b[cur][j].v,
                                                       (short)0, acc[j], false, false);
  }
  const int nloc = lane & 15;
  const int mloc = (lane >> 4) * 8;             // C layout: VGPR v -> M = v (+8 for hi lanes)
#pragma unroll
  for (int j = 0; j < 4; ++j)
#pragma unroll
    for (int v = 0; v < 8; ++v)
      C[(size_t)(mbase + mloc + v) * ldc + nbase + j * 16 + nloc] = acc[j][v];
}

// ---------------- conversion kernels ----------------
__global__ void conv_bf16_T128(const float* __restrict__ w, unsigned short* __restrict__ o) {
  int idx = blockIdx.x * blockDim.x + threadIdx.x;     // 16384 elements
  int k = idx >> 7, n = idx & 127;
  o[n * DIM + k] = f2bf_bits(w[idx]);                  // Bt[n][k] = W[k][n]  (for x @ W)
}
__global__ void conv_bf16(const float* __restrict__ w, unsigned short* __restrict__ o, int n) {
  int i = blockIdx.x * blockDim.x + threadIdx.x;
  if (i < n) o[i] = f2bf_bits(w[i]);                   // already [n][k]     (for x @ W.T)
}
// activation f32 -> bf16, 4 elements per thread (b128 in, b64 out)
__global__ void conv_act_bf16(const float* __restrict__ in, unsigned short* __restrict__ out,
                              int n4) {
  int i = blockIdx.x * blockDim.x + threadIdx.x;
  if (i >= n4) return;
  float4 v = ((const float4*)in)[i];
  ushort4 o = {f2bf_bits(v.x), f2bf_bits(v.y), f2bf_bits(v.z), f2bf_bits(v.w)};
  ((ushort4*)out)[i] = o;
}

// ---------------- small helpers ----------------
__global__ void zero_f32(float* __restrict__ p, int n) {
  int i = blockIdx.x * blockDim.x + threadIdx.x;
  int stride = gridDim.x * blockDim.x;
  for (; i < n; i += stride) p[i] = 0.f;
}
__global__ void copy_pad(const float* __restrict__ s, float* __restrict__ d, int nvalid, int ntotal) {
  int i = blockIdx.x * blockDim.x + threadIdx.x;
  if (i < ntotal) d[i] = (i < nvalid) ? s[i] : 0.f;
}
__global__ void add_elem(const float* __restrict__ a, const float* __restrict__ b,
                         float* __restrict__ c, int n) {
  int i = blockIdx.x * blockDim.x + threadIdx.x;
  if (i < n) c[i] = a[i] + b[i];
}

// ---------------- row-wise l2 normalize (one wave per 128-dim row) ----------------
__global__ void l2norm_rows(const float* __restrict__ in, float* __restrict__ out, int nrows) {
  int row = blockIdx.x * 8 + (threadIdx.x >> 5);
  int lane = threadIdx.x & 31;
  if (row >= nrows) return;
  size_t o = (size_t)row * DIM + lane * 4;
  float4 v = *(const float4*)(in + o);
  float ss = wave_sum(v.x * v.x + v.y * v.y + v.z * v.z + v.w * v.w);
  float inv = 1.0f / fmaxf(sqrtf(ss), 1e-12f);
  float4 ov = {v.x * inv, v.y * inv, v.z * inv, v.w * inv};
  *(float4*)(out + o) = ov;
}

// ---------------- relation evolution ----------------
__global__ void rel_accum(const float* __restrict__ h, const int* __restrict__ src,
                          const int* __restrict__ et, float* __restrict__ rsum,
                          float* __restrict__ rcnt, int nE) {
  int e = blockIdx.x * 8 + (threadIdx.x >> 5);
  int lane = threadIdx.x & 31;
  if (e >= nE) return;
  int s = src[e], tt = et[e];
  float4 v = *(const float4*)(h + (size_t)s * DIM + lane * 4);
  float* p = rsum + (size_t)tt * DIM + lane * 4;
  atomicAdd(p + 0, v.x); atomicAdd(p + 1, v.y);
  atomicAdd(p + 2, v.z); atomicAdd(p + 3, v.w);
  if (lane == 0) atomicAdd(rcnt + tt, 1.0f);
}
__global__ void rel_mean(const float* __restrict__ rsum, const float* __restrict__ rcnt,
                         float* __restrict__ rin) {
  int i = blockIdx.x * blockDim.x + threadIdx.x;
  if (i >= RPAD * DIM) return;
  rin[i] = rsum[i] / fmaxf(rcnt[i >> 7], 1.0f);
}
__global__ void gru_gate(const float* __restrict__ gi, const float* __restrict__ gh,
                         const float* __restrict__ bih, const float* __restrict__ bhh,
                         float* __restrict__ r) {
  int idx = blockIdx.x * blockDim.x + threadIdx.x;     // NREL*DIM
  if (idx >= NREL * DIM) return;
  int row = idx >> 7, j = idx & 127;
  const float* gir = gi + (size_t)row * 384;
  const float* ghr = gh + (size_t)row * 384;
  float i_r = gir[j] + bih[j],             h_r = ghr[j] + bhh[j];
  float i_z = gir[j + 128] + bih[j + 128], h_z = ghr[j + 128] + bhh[j + 128];
  float i_n = gir[j + 256] + bih[j + 256], h_n = ghr[j + 256] + bhh[j + 256];
  float rr = 1.0f / (1.0f + expf(-(i_r + h_r)));
  float z  = 1.0f / (1.0f + expf(-(i_z + h_z)));
  float n  = tanhf(i_n + rr * h_n);
  r[idx] = (1.0f - z) * n + z * r[idx];
}

// ---------------- RGCN message passing ----------------
// One wave per edge: 32 lanes x float4 = 128 dims; msg = hW[src] + rW[etype]
__global__ void edge_scatter(const float* __restrict__ hW, const float* __restrict__ rW,
                             const int* __restrict__ src, const int* __restrict__ dst,
                             const int* __restrict__ et, float* __restrict__ agg,
                             float* __restrict__ deg, int nE) {
  int e = blockIdx.x * 8 + (threadIdx.x >> 5);
  int lane = threadIdx.x & 31;
  if (e >= nE) return;
  int s = src[e], d = dst[e], tt = et[e];
  float4 a = *(const float4*)(hW + (size_t)s * DIM + lane * 4);
  float4 b = *(const float4*)(rW + (size_t)tt * DIM + lane * 4);
  float* p = agg + (size_t)d * DIM + lane * 4;
  atomicAdd(p + 0, a.x + b.x); atomicAdd(p + 1, a.y + b.y);
  atomicAdd(p + 2, a.z + b.z); atomicAdd(p + 3, a.w + b.w);
  if (lane == 0) atomicAdd(deg + d, 1.0f);
}
__global__ void rgcn_finalize(float* __restrict__ agg, const float* __restrict__ deg,
                              const float* __restrict__ hl, int n) {
  int i = blockIdx.x * blockDim.x + threadIdx.x;       // NENT*DIM
  if (i >= n) return;
  float x = agg[i] / fmaxf(deg[i >> 7], 1.0f) + hl[i];
  agg[i] = (x >= 0.0f) ? x : x * RRELU_SLOPE;
}
__global__ void timegate(const float* __restrict__ g, const float* __restrict__ b,
                         const float* __restrict__ cur, float* __restrict__ h, int n) {
  int i = blockIdx.x * blockDim.x + threadIdx.x;
  if (i >= n) return;
  float u = 1.0f / (1.0f + expf(-(g[i] + b[i & 127])));
  h[i] = u * cur[i] + (1.0f - u) * h[i];
}

// ---------------- fused softmax(dim=-1) * ent, then l2norm ----------------
__global__ void softmax_l2(const float* __restrict__ zr, const float* __restrict__ bias,
                           const float* __restrict__ ent, float* __restrict__ out, int nrows) {
  int row = blockIdx.x * 8 + (threadIdx.x >> 5);
  int lane = threadIdx.x & 31;
  if (row >= nrows) return;
  size_t o = (size_t)row * DIM + lane * 4;
  float4 z = *(const float4*)(zr + o);
  float4 b = *(const float4*)(bias + lane * 4);
  z.x += b.x; z.y += b.y; z.z += b.z; z.w += b.w;
  float mx = wave_max(fmaxf(fmaxf(z.x, z.y), fmaxf(z.z, z.w)));
  float ex = expf(z.x - mx), ey = expf(z.y - mx), ez = expf(z.z - mx), ew = expf(z.w - mx);
  float inv = 1.0f / wave_sum(ex + ey + ez + ew);
  float4 e = *(const float4*)(ent + o);
  float yx = ex * inv * e.x, yy = ey * inv * e.y, yz = ez * inv * e.z, yw = ew * inv * e.w;
  float ss = wave_sum(yx * yx + yy * yy + yz * yz + yw * yw);
  float invn = 1.0f / fmaxf(sqrtf(ss), 1e-12f);
  float4 ov = {yx * invn, yy * invn, yz * invn, yw * invn};
  *(float4*)(out + o) = ov;
}

extern "C" void kernel_launch(void* const* d_in, const int* in_sizes, int n_in,
                              void* d_out, int out_size, void* d_ws, size_t ws_size,
                              hipStream_t stream) {
  const float* dynamic_emb = (const float*)d_in[0];
  const float* emb_rel     = (const float*)d_in[1];
  const float* W_neigh     = (const float*)d_in[2];   // [2,128,128]
  const float* W_loop      = (const float*)d_in[3];   // [2,128,128]
  const float* time_gate_w = (const float*)d_in[4];
  const float* time_gate_b = (const float*)d_in[5];
  const float* gru_w_ih    = (const float*)d_in[6];   // [384,128]
  const float* gru_w_hh    = (const float*)d_in[7];   // [384,128]
  const float* gru_b_ih    = (const float*)d_in[8];
  const float* gru_b_hh    = (const float*)d_in[9];
  const float* w5_w        = (const float*)d_in[10];
  const float* w5_b        = (const float*)d_in[11];
  const float* sub_W_neigh = (const float*)d_in[12];
  const float* sub_W_loop  = (const float*)d_in[13];
  const float* query_mask  = (const float*)d_in[14];
  const int* src       = (const int*)d_in[15];        // [3,E]
  const int* dst       = (const int*)d_in[16];
  const int* etype     = (const int*)d_in[17];
  const int* sub_src   = (const int*)d_in[18];
  const int* sub_dst   = (const int*)d_in[19];
  const int* sub_etype = (const int*)d_in[20];

  // ---- workspace carve-out ----
  char* base = (char*)d_ws;
  size_t off = 0;
  auto alloc = [&](size_t bytes) -> void* {
    void* p = base + off;
    off += (bytes + 255) & ~(size_t)255;
    return p;
  };
  unsigned short* wnT0 = (unsigned short*)alloc(DIM * DIM * 2);
  unsigned short* wnT1 = (unsigned short*)alloc(DIM * DIM * 2);
  unsigned short* wlT0 = (unsigned short*)alloc(DIM * DIM * 2);
  unsigned short* wlT1 = (unsigned short*)alloc(DIM * DIM * 2);
  unsigned short* tgT  = (unsigned short*)alloc(DIM * DIM * 2);
  unsigned short* swnT = (unsigned short*)alloc(DIM * DIM * 2);
  unsigned short* swlT = (unsigned short*)alloc(DIM * DIM * 2);
  unsigned short* w5B  = (unsigned short*)alloc(DIM * DIM * 2);
  unsigned short* wihB = (unsigned short*)alloc(384 * DIM * 2);
  unsigned short* whhB = (unsigned short*)alloc(384 * DIM * 2);
  unsigned short* rB    = (unsigned short*)alloc((size_t)RPAD * DIM * 2);
  unsigned short* rinB  = (unsigned short*)alloc((size_t)RPAD * DIM * 2);
  unsigned short* hisrB = (unsigned short*)alloc((size_t)RPAD * DIM * 2);
  unsigned short* XB    = (unsigned short*)alloc((size_t)NENT * DIM * 2);  // bf16 activations
  float* r    = (float*)alloc((size_t)RPAD * DIM * 4);
  float* rsum = (float*)alloc((size_t)RPAD * DIM * 4);
  float* rcnt = (float*)alloc((size_t)RPAD * 4);
  float* rin  = (float*)alloc((size_t)RPAD * DIM * 4);
  float* rW   = (float*)alloc((size_t)RPAD * DIM * 4);
  float* hisr = (float*)alloc((size_t)RPAD * DIM * 4);
  float* gi   = (float*)alloc((size_t)RPAD * 384 * 4);
  float* gh   = (float*)alloc((size_t)RPAD * 384 * 4);
  float* deg  = (float*)alloc((size_t)NENT * 4);
  float* B0 = (float*)alloc((size_t)NENT * DIM * 4);  // h
  float* B1 = (float*)alloc((size_t)NENT * DIM * 4);  // agg -> cur (in place)
  float* B2 = (float*)alloc((size_t)NENT * DIM * 4);  // hW / gate matmul out
  float* B3 = (float*)alloc((size_t)NENT * DIM * 4);  // hL / (q+ent)

  auto convA = [&](const float* f, unsigned short* b, int nelem) {
    conv_act_bf16<<<(nelem / 4 + 255) / 256, 256, 0, stream>>>(f, b, nelem / 4);
  };
  auto matmul = [&](const unsigned short* A, const unsigned short* Bt, float* C,
                    int M, int ncols) {
    dim3 grid((M / 16 + 7) / 8, ncols / 64);           // 8 mtile-waves per block
    mm_bf16_wmma<<<grid, 256, 0, stream>>>(A, Bt, C, M / 16, ncols);
  };
  const int ND = NENT * DIM;

  // ---- prep: weight conversion to bf16 [n][k] layout ----
  conv_bf16_T128<<<64, 256, 0, stream>>>(W_neigh, wnT0);
  conv_bf16_T128<<<64, 256, 0, stream>>>(W_neigh + DIM * DIM, wnT1);
  conv_bf16_T128<<<64, 256, 0, stream>>>(W_loop, wlT0);
  conv_bf16_T128<<<64, 256, 0, stream>>>(W_loop + DIM * DIM, wlT1);
  conv_bf16_T128<<<64, 256, 0, stream>>>(time_gate_w, tgT);
  conv_bf16_T128<<<64, 256, 0, stream>>>(sub_W_neigh, swnT);
  conv_bf16_T128<<<64, 256, 0, stream>>>(sub_W_loop, swlT);
  conv_bf16<<<64, 256, 0, stream>>>(w5_w, w5B, DIM * DIM);        // used as x @ W.T
  conv_bf16<<<192, 256, 0, stream>>>(gru_w_ih, wihB, 384 * DIM);  // used as x @ W.T
  conv_bf16<<<192, 256, 0, stream>>>(gru_w_hh, whhB, 384 * DIM);

  // ---- init: h = l2norm(dynamic_emb); r = emb_rel (padded to 512 rows) ----
  l2norm_rows<<<(NENT + 7) / 8, 256, 0, stream>>>(dynamic_emb, B0, NENT);
  copy_pad<<<(RPAD * DIM + 255) / 256, 256, 0, stream>>>(emb_rel, r, NREL * DIM, RPAD * DIM);

  for (int t = 0; t < TSTEPS; ++t) {
    const int* s_t = src + t * NEDGE;
    const int* d_t = dst + t * NEDGE;
    const int* e_t = etype + t * NEDGE;
    // relation evolution: per-relation mean of source states -> GRU cell
    zero_f32<<<64, 256, 0, stream>>>(rsum, RPAD * DIM);
    zero_f32<<<2, 256, 0, stream>>>(rcnt, RPAD);
    rel_accum<<<NEDGE / 8, 256, 0, stream>>>(B0, s_t, e_t, rsum, rcnt, NEDGE);
    rel_mean<<<(RPAD * DIM + 255) / 256, 256, 0, stream>>>(rsum, rcnt, rin);
    convA(rin, rinB, RPAD * DIM);
    convA(r, rB, RPAD * DIM);                          // old r for gh
    matmul(rinB, wihB, gi, RPAD, 384);   // WMMA
    matmul(rB,   whhB, gh, RPAD, 384);   // WMMA
    gru_gate<<<(NREL * DIM + 255) / 256, 256, 0, stream>>>(gi, gh, gru_b_ih, gru_b_hh, r);
    convA(r, rB, RPAD * DIM);                          // evolved r for rW matmuls
    // 2 RGCN layers: msg = hW[src] + rW[etype]  (matmul distributed over the add)
    for (int l = 0; l < 2; ++l) {
      const float* X = (l == 0) ? B0 : B1;
      const unsigned short* WnT = (l == 0) ? wnT0 : wnT1;
      const unsigned short* WlT = (l == 0) ? wlT0 : wlT1;
      convA(X, XB, ND);
      matmul(XB, WnT, B2, NENT, DIM);   // WMMA: hW
      matmul(XB, WlT, B3, NENT, DIM);   // WMMA: self-loop
      matmul(rB, WnT, rW, RPAD, DIM);   // WMMA: relation term
      zero_f32<<<4096, 256, 0, stream>>>(B1, ND);
      zero_f32<<<512, 256, 0, stream>>>(deg, NENT);
      edge_scatter<<<NEDGE / 8, 256, 0, stream>>>(B2, rW, s_t, d_t, e_t, B1, deg, NEDGE);
      rgcn_finalize<<<(ND + 255) / 256, 256, 0, stream>>>(B1, deg, B3, ND);
    }
    l2norm_rows<<<(NENT + 7) / 8, 256, 0, stream>>>(B1, B1, NENT);
    convA(B1, XB, ND);
    matmul(XB, tgT, B2, NENT, DIM);     // WMMA: time gate
    timegate<<<(ND + 255) / 256, 256, 0, stream>>>(B2, time_gate_b, B1, B0, ND);
  }

  // ---- history encoding on union subgraph ----
  l2norm_rows<<<RPAD / 8, 256, 0, stream>>>(r, hisr, RPAD);
  convA(hisr, hisrB, RPAD * DIM);
  convA(B0, XB, ND);
  matmul(XB, swnT, B2, NENT, DIM);
  matmul(XB, swlT, B3, NENT, DIM);
  matmul(hisrB, swnT, rW, RPAD, DIM);
  zero_f32<<<4096, 256, 0, stream>>>(B1, ND);
  zero_f32<<<512, 256, 0, stream>>>(deg, NENT);
  edge_scatter<<<NSUB / 8, 256, 0, stream>>>(B2, rW, sub_src, sub_dst, sub_etype, B1, deg, NSUB);
  rgcn_finalize<<<(ND + 255) / 256, 256, 0, stream>>>(B1, deg, B3, ND);   // B1 = his_ent
  // attention + output
  add_elem<<<(ND + 255) / 256, 256, 0, stream>>>(query_mask, B1, B3, ND); // q + his_ent
  convA(B3, XB, ND);
  matmul(XB, w5B, B2, NENT, DIM);      // WMMA: (q+ent) @ w5.T
  softmax_l2<<<(NENT + 7) / 8, 256, 0, stream>>>(B2, w5_b, B1, (float*)d_out, NENT);
}
